// RCNNNet_24283745092095
// MI455X (gfx1250) — matrix operand order repaired and checked by reading, not compiled, not run on hardware
//
#include <hip/hip_runtime.h>

// ---------------- CDNA5 WMMA types ----------------
typedef __attribute__((ext_vector_type(16))) __bf16 v16bf;
typedef __attribute__((ext_vector_type(2)))  __bf16 v2bf;
typedef __attribute__((ext_vector_type(8)))  float  v8f;
typedef __attribute__((ext_vector_type(2)))  float  v2f;

union BFrag {
    v16bf          v;
    unsigned short us[16];
    unsigned int   ui[8];
};

// Scalar f32 -> bf16 (prep kernel only).
__device__ __forceinline__ unsigned short f2bf(float f) {
    union { __bf16 h; unsigned short u; } cv;
    cv.h = (__bf16)f;
    return cv.u;
}

// Packed pair conversion -> single v_cvt_pk_bf16_f32.
__device__ __forceinline__ unsigned int pack_bf(float lo, float hi) {
    union { v2bf v; unsigned int u; } cv;
#if __has_builtin(__builtin_amdgcn_cvt_pk_bf16_f32)
    cv.v = __builtin_amdgcn_cvt_pk_bf16_f32(lo, hi);
#else
    v2f t = {lo, hi};
    cv.v = __builtin_convertvector(t, v2bf);   // vector fptrunc -> packed cvt
#endif
    return cv.u;
}

__device__ __forceinline__ v8f wmma_bf16(const BFrag& a, const BFrag& b, v8f c) {
    return __builtin_amdgcn_wmma_f32_16x16x32_bf16(
        /*neg_a=*/false, a.v, /*neg_b=*/false, b.v,
        /*c_mod=*/(short)0, c, /*reuse_a=*/false, /*reuse_b=*/false);
}

// A-fragment (16x32 bf16, M x K) from row-major bf16 weight [Mdim][Kdim].
// lanes 0-15 & 16-31 both hold M = mt*16 + ncol; K pattern (contiguous k = half*16 + s, same order as the B fragments):
//   regs 0..3 : K = half*16 + 2j
//   regs 4..7 : K = half*16 + 8 + 2j
__device__ __forceinline__ BFrag load_afrag(const unsigned short* __restrict__ w,
                                            int Kdim, int mt, int ks,
                                            int ncol, int half) {
    BFrag a;
    const unsigned int* row32 =
        (const unsigned int*)(w + (size_t)(mt * 16 + ncol) * Kdim + ks * 32);
#pragma unroll
    for (int j = 0; j < 4; ++j) a.ui[j]     = row32[half * 8 + j];
#pragma unroll
    for (int j = 0; j < 4; ++j) a.ui[4 + j] = row32[half * 8 + 4 + j];
    return a;
}

// ---------------- problem constants ----------------
#define Bb   64
#define ICn  32
#define PCn  128
#define RCn  32
#define OUTn 128
#define Hh   96
#define Ww   320
#define NPTS 2048

// ---------------- weight prep: f32 -> bf16, BN folding ----------------
__global__ void prep_kernel(const float* __restrict__ Wf1, const float* __restrict__ Wf2,
                            const float* __restrict__ Wia, const float* __restrict__ Wfuse,
                            const float* __restrict__ b_ia,
                            const float* __restrict__ g1, const float* __restrict__ bt1,
                            const float* __restrict__ m1, const float* __restrict__ v1,
                            const float* __restrict__ b_fuse,
                            const float* __restrict__ g2, const float* __restrict__ bt2,
                            const float* __restrict__ m2, const float* __restrict__ v2,
                            unsigned short* __restrict__ wf1b, unsigned short* __restrict__ wf2b,
                            unsigned short* __restrict__ wiab, unsigned short* __restrict__ wfub,
                            float* __restrict__ scale1, float* __restrict__ shift1,
                            float* __restrict__ scale2, float* __restrict__ shift2) {
    int tid = blockIdx.x * blockDim.x + threadIdx.x;
    int stride = gridDim.x * blockDim.x;
    for (int i = tid; i < RCn * ICn;    i += stride) wf1b[i] = f2bf(Wf1[i]);
    for (int i = tid; i < RCn * PCn;    i += stride) wf2b[i] = f2bf(Wf2[i]);
    for (int i = tid; i < PCn * ICn;    i += stride) wiab[i] = f2bf(Wia[i]);
    for (int i = tid; i < OUTn * 2*PCn; i += stride) wfub[i] = f2bf(Wfuse[i]);
    for (int c = tid; c < PCn; c += stride) {
        float s1 = g1[c] * rsqrtf(v1[c] + 1e-5f);
        scale1[c] = s1;
        shift1[c] = (b_ia[c] - m1[c]) * s1 + bt1[c];
        float s2 = g2[c] * rsqrtf(v2[c] + 1e-5f);
        scale2[c] = s2;
        shift2[c] = (b_fuse[c] - m2[c]) * s2 + bt2[c];
    }
}

// ---------------- fused gather + attention + conv kernel ----------------
// block = 256 threads = 8 waves; each wave owns a 16-point tile of one batch.
// grid = B * (N/128) = 1024 blocks.
__global__ __launch_bounds__(256) void fuse_kernel(
    const float* __restrict__ img_map, const float* __restrict__ xy,
    const float* __restrict__ pf,
    const unsigned short* __restrict__ wf1b, const unsigned short* __restrict__ wf2b,
    const unsigned short* __restrict__ wiab, const unsigned short* __restrict__ wfub,
    const float* __restrict__ bf1, const float* __restrict__ bf2,
    const float* __restrict__ Wf3, const float* __restrict__ bf3,
    const float* __restrict__ scale1, const float* __restrict__ shift1,
    const float* __restrict__ scale2, const float* __restrict__ shift2,
    float* __restrict__ out) {

    // per-wave img_new staged as packed bf16x2 dwords: word = ch/2, 16 points wide
    __shared__ unsigned int lds_new[8 * (PCn / 2) * 16];   // 32 KB

    const int lane = threadIdx.x & 31;
    const int wave = threadIdx.x >> 5;
    const int half = lane >> 4;
    const int ncol = lane & 15;
    const int b    = blockIdx.x >> 4;          // 16 segments per batch
    const int seg  = blockIdx.x & 15;
    const int p    = seg * 128 + wave * 16 + ncol;   // this lane's point

    // ---- bilinear gather of 32 image channels -> B-fragment (K=32 x N=16)
    // lane covers channels c = half*16 + t  (== the fragment's K pattern)
    float fx = xy[((size_t)b * NPTS + p) * 2 + 0];
    float fy = xy[((size_t)b * NPTS + p) * 2 + 1];
    float ix = ((fx + 1.f) * (float)Ww - 1.f) * 0.5f;
    float iy = ((fy + 1.f) * (float)Hh - 1.f) * 0.5f;
    float ix0f = floorf(ix), iy0f = floorf(iy);
    int ix0 = (int)ix0f, iy0 = (int)iy0f;
    int ix1 = ix0 + 1,   iy1 = iy0 + 1;
    float wx1 = ix - ix0f, wx0 = 1.f - wx1;
    float wy1 = iy - iy0f, wy0 = 1.f - wy1;
    bool vx0 = (ix0 >= 0) & (ix0 < Ww), vx1 = (ix1 >= 0) & (ix1 < Ww);
    bool vy0 = (iy0 >= 0) & (iy0 < Hh), vy1 = (iy1 >= 0) & (iy1 < Hh);
    float w00 = (vx0 & vy0) ? wx0 * wy0 : 0.f;
    float w10 = (vx1 & vy0) ? wx1 * wy0 : 0.f;
    float w01 = (vx0 & vy1) ? wx0 * wy1 : 0.f;
    float w11 = (vx1 & vy1) ? wx1 * wy1 : 0.f;
    int cx0 = ix0 < 0 ? 0 : (ix0 > Ww - 1 ? Ww - 1 : ix0);
    int cx1 = ix1 < 0 ? 0 : (ix1 > Ww - 1 ? Ww - 1 : ix1);
    int cy0 = iy0 < 0 ? 0 : (iy0 > Hh - 1 ? Hh - 1 : iy0);
    int cy1 = iy1 < 0 ? 0 : (iy1 > Hh - 1 ? Hh - 1 : iy1);
    const size_t o00 = (size_t)cy0 * Ww + cx0, o10 = (size_t)cy0 * Ww + cx1;
    const size_t o01 = (size_t)cy1 * Ww + cx0, o11 = (size_t)cy1 * Ww + cx1;

    BFrag bimg;
    {
        const float* base = img_map + ((size_t)b * ICn + half * 16) * (Hh * Ww);
#pragma unroll
        for (int t = 0; t < 16; t += 2) {
            const float* mp0 = base + (size_t)t * (Hh * Ww);
            const float* mp1 = mp0 + (Hh * Ww);
            float v0 = w00 * mp0[o00] + w10 * mp0[o10] + w01 * mp0[o01] + w11 * mp0[o11];
            float v1 = w00 * mp1[o00] + w10 * mp1[o10] + w01 * mp1[o01] + w11 * mp1[o11];
            bimg.ui[t >> 1] = pack_bf(v0, v1);
        }
    }

    // ---- point-feature B-fragments for K=128 (kept in regs, reused by fuse)
    BFrag bpf[4];
#pragma unroll
    for (int ks = 0; ks < 4; ++ks) {
#pragma unroll
        for (int j = 0; j < 8; ++j) {
            int k = ks * 32 + half * 16 + 2 * j;
            float lo = pf[((size_t)b * PCn + k)     * NPTS + p];
            float hi = pf[((size_t)b * PCn + k + 1) * NPTS + p];
            bpf[ks].ui[j] = pack_bf(lo, hi);
        }
    }

    // ---- attention gate: ri + rp (32 x 16) via WMMA, then tanh/dot/sigmoid
    float att;
    {
        v8f acc[2];
#pragma unroll
        for (int mt = 0; mt < 2; ++mt) {
#pragma unroll
            for (int j = 0; j < 8; ++j) {
                int r = mt * 16 + half * 8 + j;
                acc[mt][j] = bf1[r] + bf2[r];
            }
            BFrag a1 = load_afrag(wf1b, ICn, mt, 0, ncol, half);
            acc[mt] = wmma_bf16(a1, bimg, acc[mt]);
#pragma unroll
            for (int ks = 0; ks < 4; ++ks) {
                BFrag a2 = load_afrag(wf2b, PCn, mt, ks, ncol, half);
                acc[mt] = wmma_bf16(a2, bpf[ks], acc[mt]);
            }
        }
        float partial = 0.f;
#pragma unroll
        for (int mt = 0; mt < 2; ++mt)
#pragma unroll
            for (int j = 0; j < 8; ++j) {
                int r = mt * 16 + half * 8 + j;
                partial += Wf3[r] * tanhf(acc[mt][j]);
            }
        partial += __shfl_xor(partial, 16, 32);   // combine the two lane-halves (same point)
        att = 1.f / (1.f + __expf(-(partial + bf3[0])));
    }

    // ---- img_new = relu(BN(Wia @ img)) * att -> packed bf16x2 dwords in LDS
    unsigned int* myl = lds_new + wave * ((PCn / 2) * 16);
#pragma unroll
    for (int mt = 0; mt < 8; ++mt) {
        v8f c = {0.f, 0.f, 0.f, 0.f, 0.f, 0.f, 0.f, 0.f};
        BFrag a = load_afrag(wiab, ICn, mt, 0, ncol, half);
        c = wmma_bf16(a, bimg, c);
#pragma unroll
        for (int j = 0; j < 8; j += 2) {
            int ch = mt * 16 + half * 8 + j;      // even channel of the pair
            float v0 = fmaxf(c[j]     * scale1[ch]     + shift1[ch],     0.f) * att;
            float v1 = fmaxf(c[j + 1] * scale1[ch + 1] + shift1[ch + 1], 0.f) * att;
            myl[(ch >> 1) * 16 + ncol] = pack_bf(v0, v1);  // one ds_store_b32
        }
    }
    __syncthreads();

    // ---- re-fragment img_new from LDS as B operand (channels = K), packed reads
    BFrag bnew[4];
#pragma unroll
    for (int s = 0; s < 4; ++s) {
#pragma unroll
        for (int j = 0; j < 8; ++j) {
            int word = s * 16 + half * 8 + j;     // = (s*32 + half*16 + 2j) / 2
            bnew[s].ui[j] = myl[word * 16 + ncol]; // one ds_load_b32, no repacking
        }
    }

    // ---- fuse: out = relu(BN(Wfuse @ [pf ; img_new]))   (128 x 16, K = 256)
#pragma unroll
    for (int mt = 0; mt < 8; ++mt) {
        v8f c = {0.f, 0.f, 0.f, 0.f, 0.f, 0.f, 0.f, 0.f};
#pragma unroll
        for (int ks = 0; ks < 8; ++ks) {
            BFrag a = load_afrag(wfub, 2 * PCn, mt, ks, ncol, half);
            c = wmma_bf16(a, (ks < 4) ? bpf[ks] : bnew[ks - 4], c);
        }
#pragma unroll
        for (int j = 0; j < 8; ++j) {
            int ch = mt * 16 + half * 8 + j;
            float v = c[j] * scale2[ch] + shift2[ch];
            out[((size_t)b * OUTn + ch) * NPTS + p] = fmaxf(v, 0.f);
        }
    }
}

// ---------------- host launch ----------------
extern "C" void kernel_launch(void* const* d_in, const int* in_sizes, int n_in,
                              void* d_out, int out_size, void* d_ws, size_t ws_size,
                              hipStream_t stream) {
    const float* img_map = (const float*)d_in[0];
    const float* xy      = (const float*)d_in[1];
    const float* pf      = (const float*)d_in[2];
    const float* W_ia    = (const float*)d_in[3];
    const float* b_ia    = (const float*)d_in[4];
    const float* g1      = (const float*)d_in[5];
    const float* bt1     = (const float*)d_in[6];
    const float* m1      = (const float*)d_in[7];
    const float* v1      = (const float*)d_in[8];
    const float* Wf1     = (const float*)d_in[9];
    const float* bf1     = (const float*)d_in[10];
    const float* Wf2     = (const float*)d_in[11];
    const float* bf2     = (const float*)d_in[12];
    const float* Wf3     = (const float*)d_in[13];
    const float* bf3     = (const float*)d_in[14];
    const float* W_fuse  = (const float*)d_in[15];
    const float* b_fuse  = (const float*)d_in[16];
    const float* g2      = (const float*)d_in[17];
    const float* bt2     = (const float*)d_in[18];
    const float* m2      = (const float*)d_in[19];
    const float* v2      = (const float*)d_in[20];

    char* ws = (char*)d_ws;
    unsigned short* wf1b   = (unsigned short*)(ws + 0);       //  2048 B
    unsigned short* wf2b   = (unsigned short*)(ws + 2048);    //  8192 B
    unsigned short* wiab   = (unsigned short*)(ws + 10240);   //  8192 B
    unsigned short* wfub   = (unsigned short*)(ws + 18432);   // 65536 B
    float*          scale1 = (float*)(ws + 83968);
    float*          shift1 = (float*)(ws + 84480);
    float*          scale2 = (float*)(ws + 84992);
    float*          shift2 = (float*)(ws + 85504);

    prep_kernel<<<64, 256, 0, stream>>>(Wf1, Wf2, W_ia, W_fuse,
                                        b_ia, g1, bt1, m1, v1,
                                        b_fuse, g2, bt2, m2, v2,
                                        wf1b, wf2b, wiab, wfub,
                                        scale1, shift1, scale2, shift2);

    fuse_kernel<<<Bb * (NPTS / 128), 256, 0, stream>>>(
        img_map, xy, pf, wf1b, wf2b, wiab, wfub,
        bf1, bf2, Wf3, bf3, scale1, shift1, scale2, shift2,
        (float*)d_out);
}
